// ConvKNRM_40063454937098
// MI455X (gfx1250) — matrix operand order repaired and checked by reading, not compiled, not run on hardware
//
#include <hip/hip_runtime.h>
#include <hip/hip_bf16.h>
#include <math.h>

// ---------------------------------------------------------------------------
// Conv-KNRM forward for MI455X (gfx1250, wave32, WMMA).
//   B=32, QLEN=32, DLEN=1024, EMBED=300, KDIM=128, KNUM=21, KSIZES={1,2,3}
// Conv AND similarity as implicit GEMMs with v_wmma_f32_16x16x32_f16;
// Gaussian kernel pooling on VALU/trans units (v_exp_f32).
// ---------------------------------------------------------------------------

#define BATCH   32
#define QLEN    32
#define DLEN    1024
#define EMBED   300
#define EPAD    320      // 10 * 32, padded embedding stride (halves)
#define KDIM    128
#define KNUM    21
#define NPAIR   9
#define FEAT    (KNUM * NPAIR)   // 189

typedef __attribute__((ext_vector_type(16))) _Float16 v16h;
typedef __attribute__((ext_vector_type(8)))  _Float16 half8;
typedef __attribute__((ext_vector_type(8)))  float    v8f;

// ---------------------------------------------------------------------------
// 1) Gather embeddings -> f16, row stride EPAD, zero pad cols/rows.
//    X layout: [B][Lalloc][EPAD] f16.  grid=(Lalloc, B), block=EPAD threads.
// ---------------------------------------------------------------------------
__global__ __launch_bounds__(EPAD)
void gather_emb_f16(const int* __restrict__ ids, const float* __restrict__ emb,
                    _Float16* __restrict__ X, int Lvalid, int Lalloc) {
    const int l = blockIdx.x;
    const int b = blockIdx.y;
    const int e = threadIdx.x;
    float v = 0.0f;
    if (l < Lvalid && e < EMBED) {
        const int id = ids[b * Lvalid + l];
        v = emb[(size_t)id * EMBED + e];
    }
    X[((size_t)b * Lalloc + l) * EPAD + e] = (_Float16)v;
}

// ---------------------------------------------------------------------------
// 2) Pack conv weights (O,E,K) fp32 -> Wt[o][KPAD] f16, K-major per channel,
//    zero stripes matching the EPAD layout.  KPAD = EPAD * k.
// ---------------------------------------------------------------------------
__global__ __launch_bounds__(256)
void pack_weights(const float* __restrict__ w, _Float16* __restrict__ Wt, int k) {
    const int KPAD  = EPAD * k;
    const int total = KDIM * KPAD;
    const int idx   = blockIdx.x * 256 + threadIdx.x;
    if (idx >= total) return;
    const int o  = idx / KPAD;
    const int j  = idx % KPAD;
    const int kk = j / EPAD;
    const int e  = j % EPAD;
    float v = 0.0f;
    if (e < EMBED) v = w[(size_t)o * EMBED * k + e * k + kk];
    Wt[(size_t)o * KPAD + j] = (_Float16)v;
}

// ---------------------------------------------------------------------------
// 3) Conv + bias + ReLU + L2-row-normalize via WMMA implicit GEMM.
//    grid = (LpadE/16, B), block = 256 (8 waves; wave w owns channels 16w..).
//    A fragment (16-bit 16x32, ISA 7.12.2): lane M=lane&15,
//      koff = (lane<16 ? 0 : 8); v[0..7]=A[M][kb+koff .. +7],
//      v[8..15]=A[M][kb+16+koff .. +7].
//    B fragment (32x16): lane N=lane&15, same K order as A:
//      v[0..7]=B[kb+koff..+7][N], v[8..15]=B[kb+16+koff..+7][N].
// ---------------------------------------------------------------------------
__global__ __launch_bounds__(256)
void conv_enc_wmma(const _Float16* __restrict__ X, int LallocX,
                   const _Float16* __restrict__ Wt, int KPAD,
                   const float* __restrict__ bias,
                   _Float16* __restrict__ ENC, int LpadE) {
    __shared__ float tile[16 * KDIM];
    __shared__ float scl[16];

    const int b    = blockIdx.y;
    const int m0   = blockIdx.x * 16;
    const int wave = threadIdx.x >> 5;
    const int lane = threadIdx.x & 31;
    const int hi   = (lane & 16) ? 1 : 0;

    const int M    = lane & 15;
    const int koff = hi ? 8 : 0;
    const _Float16* arow = X + ((size_t)b * LallocX + (m0 + M)) * EPAD;

    const int n0   = wave * 16;
    const int N    = lane & 15;
    const int ksel = hi ? 8 : 0;
    const _Float16* brow = Wt + (size_t)(n0 + N) * KPAD;

    v8f c = {};
    const int nsteps = KPAD / 32;
    for (int kt = 0; kt < nsteps; ++kt) {
        const int kb = kt * 32;
        half8 a0 = *(const half8*)(arow + kb + koff);
        half8 a1 = *(const half8*)(arow + kb + 16 + koff);
        half8 b0 = *(const half8*)(brow + kb + ksel);
        half8 b1 = *(const half8*)(brow + kb + 16 + ksel);
        v16h Af, Bf;
#pragma unroll
        for (int i = 0; i < 8; ++i) {
            Af[i] = a0[i]; Af[8 + i] = a1[i];
            Bf[i] = b0[i]; Bf[8 + i] = b1[i];
        }
        c = __builtin_amdgcn_wmma_f32_16x16x32_f16(
                false, Af, false, Bf, (short)0, c, false, false);
    }

    // bias + relu, scatter to LDS (C layout: VGPR r -> M = r + 8*hi, N = lane&15)
    const int   col = n0 + N;
    const float bn  = bias[col];
    const int   mb  = hi ? 8 : 0;
#pragma unroll
    for (int r = 0; r < 8; ++r) {
        float v = c[r] + bn;
        v = fmaxf(v, 0.0f);
        tile[(r + mb) * KDIM + col] = v;
    }
    __syncthreads();

    if (threadIdx.x < 16) {
        float s = 0.0f;
        const float* row = &tile[threadIdx.x * KDIM];
        for (int j = 0; j < KDIM; ++j) s += row[j] * row[j];
        scl[threadIdx.x] = 1.0f / (sqrtf(s) + 1e-13f);
    }
    __syncthreads();

    for (int idx = threadIdx.x; idx < 16 * KDIM; idx += 256) {
        const int m = idx >> 7;
        const int o = idx & 127;
        ENC[((size_t)b * LpadE + (m0 + m)) * KDIM + o] = (_Float16)(tile[idx] * scl[m]);
    }
}

// ---------------------------------------------------------------------------
// 4) Matcher with WMMA similarity: one block per batch item, one launch per
//    (qk, dk) pair.  Sim tile computed as C = D_tile(A, M=v) x Q_tile^T(B, N=q)
//    so each lane's 8 C values share a fixed q (lane) and vary in v (VGPR),
//    and the Gaussian histogram (sum over v) reduces within-lane into 21
//    per-lane accumulators.  Wave w: q-tile = w&1, v-group = w>>1 (stride 64).
//    Deterministic LDS reduction over the 8 partials per q.
// ---------------------------------------------------------------------------
__global__ __launch_bounds__(256)
void matcher_wmma(const _Float16* __restrict__ QE, const _Float16* __restrict__ DE,
                  const float* __restrict__ qmask, const float* __restrict__ dmask,
                  float* __restrict__ logits, int Qv, int Vv, int pairBase) {
    __shared__ float dm_s[DLEN];               // 4 KB
    __shared__ float Sg[8 * 32 * KNUM];        // 21504 B  per-(wave,lane) partials
    __shared__ float S2[QLEN * KNUM];          // 2688 B   per-q Gaussian sums

    const int b    = blockIdx.x;
    const int tid  = threadIdx.x;
    const int wave = tid >> 5;
    const int lane = tid & 31;
    const int hi   = (lane & 16) ? 1 : 0;
    const int qtil = wave & 1;                 // q-tile 0 or 1
    const int vgrp = wave >> 1;                // 0..3, v-tiles strided by 64

    for (int idx = tid; idx < DLEN; idx += 256)
        dm_s[idx] = dmask[b * DLEN + idx];
    __syncthreads();

    // KNRM mus / 1/(2 sig^2) — compile-time constants after unroll.
    float mu[KNUM], w2[KNUM];
    mu[0] = 1.0f; w2[0] = 500000.0f;           // sig=0.001
    { float m = 0.95f; for (int i = 1; i < KNUM; ++i) { mu[i] = m; m -= 0.1f; w2[i] = 50.0f; } }

    const int   q  = qtil * 16 + (lane & 15);  // this lane's query row
    const float qm = qmask[b * QLEN + q];

    // B fragment (K x 16 = Q_tile^T): fixed for the whole v loop; 4 K-steps.
    const _Float16* brow = QE + ((size_t)b * QLEN + q) * KDIM;
    const int ksel = hi ? 8 : 0;
    v16h Bf[4];
#pragma unroll
    for (int kt = 0; kt < 4; ++kt) {
        half8 b0 = *(const half8*)(brow + kt * 32 + ksel);
        half8 b1 = *(const half8*)(brow + kt * 32 + 16 + ksel);
#pragma unroll
        for (int i = 0; i < 8; ++i) { Bf[kt][i] = b0[i]; Bf[kt][8 + i] = b1[i]; }
    }

    float acc[KNUM];
#pragma unroll
    for (int i = 0; i < KNUM; ++i) acc[i] = 0.0f;

    const int M    = lane & 15;
    const int koff = hi ? 8 : 0;

    for (int v0 = vgrp * 16; v0 < Vv; v0 += 64) {
        const _Float16* arow = DE + ((size_t)b * DLEN + (v0 + M)) * KDIM;
        v8f c = {};
#pragma unroll
        for (int kt = 0; kt < 4; ++kt) {
            half8 a0 = *(const half8*)(arow + kt * 32 + koff);
            half8 a1 = *(const half8*)(arow + kt * 32 + 16 + koff);
            v16h Af;
#pragma unroll
            for (int i = 0; i < 8; ++i) { Af[i] = a0[i]; Af[8 + i] = a1[i]; }
            c = __builtin_amdgcn_wmma_f32_16x16x32_f16(
                    false, Af, false, Bf[kt], (short)0, c, false, false);
        }
        // C layout: VGPR r -> v = v0 + r + 8*hi, n(=q) = lane&15 within tile.
#pragma unroll
        for (int r = 0; r < 8; ++r) {
            const int v = v0 + r + 8 * hi;
            if (v < Vv) {
                const float ms = c[r] * qm * dm_s[v];
#pragma unroll
                for (int i = 0; i < KNUM; ++i) {
                    const float d = ms - mu[i];
                    acc[i] += __expf(-d * d * w2[i]);
                }
            }
        }
    }

    float* sg = &Sg[(wave * 32 + lane) * KNUM];
#pragma unroll
    for (int i = 0; i < KNUM; ++i) sg[i] = acc[i];
    __syncthreads();

    // Per-q sums over the 8 contributing (wave,lane) slots, fixed order.
    for (int idx = tid; idx < QLEN * KNUM; idx += 256) {
        const int qq = idx / KNUM;
        const int i  = idx % KNUM;
        const int qt = qq >> 4;
        const int ql = qq & 15;
        float s = 0.0f;
#pragma unroll
        for (int vg = 0; vg < 4; ++vg) {
            const int w = vg * 2 + qt;
            s += Sg[(w * 32 + ql) * KNUM + i];
            s += Sg[(w * 32 + ql + 16) * KNUM + i];
        }
        S2[idx] = s;
    }
    __syncthreads();

    if (tid < KNUM) {
        float f = 0.0f;
        for (int qq = 0; qq < Qv; ++qq)
            f += __logf(fmaxf(S2[qq * KNUM + tid], 1e-10f)) * 0.01f * qmask[b * QLEN + qq];
        logits[(size_t)b * FEAT + pairBase + tid] = f;
    }
}

// ---------------------------------------------------------------------------
// 5) Dense head: score[b] = logits[b] . w + bias
// ---------------------------------------------------------------------------
__global__ __launch_bounds__(32)
void dense_kernel(const float* __restrict__ logits, const float* __restrict__ dw,
                  const float* __restrict__ db, float* __restrict__ score) {
    const int b = threadIdx.x;
    float s = db[0];
    for (int f = 0; f < FEAT; ++f)
        s += logits[(size_t)b * FEAT + f] * dw[f];
    score[b] = s;
}

// ---------------------------------------------------------------------------
// Host launcher
// ---------------------------------------------------------------------------
extern "C" void kernel_launch(void* const* d_in, const int* in_sizes, int n_in,
                              void* d_out, int out_size, void* d_ws, size_t ws_size,
                              hipStream_t stream) {
    const int*   query_ids   = (const int*)  d_in[0];
    const float* query_masks = (const float*)d_in[1];
    const int*   doc_ids     = (const int*)  d_in[2];
    const float* doc_masks   = (const float*)d_in[3];
    const float* emb         = (const float*)d_in[4];
    const float* conv_w[3]   = { (const float*)d_in[5], (const float*)d_in[7],  (const float*)d_in[9]  };
    const float* conv_b[3]   = { (const float*)d_in[6], (const float*)d_in[8],  (const float*)d_in[10] };
    const float* dense_w     = (const float*)d_in[11];
    const float* dense_b     = (const float*)d_in[12];

    float* out_score  = (float*)d_out;
    float* out_logits = (float*)d_out + BATCH;

    // ---- workspace layout (256 B aligned segments), ~48 MB total ----
    const int QALLOC = QLEN + 2;   // 34 rows (k-1 overread + spare)
    const int DALLOC = DLEN + 2;   // 1026 rows
    char* ws = (char*)d_ws;
    size_t off = 0;
    auto take = [&](size_t bytes) { char* p = ws + off; off = (off + bytes + 255) & ~(size_t)255; return p; };

    _Float16* XQ = (_Float16*)take((size_t)BATCH * QALLOC * EPAD * 2);
    _Float16* XD = (_Float16*)take((size_t)BATCH * DALLOC * EPAD * 2);
    _Float16* WT[3];
    for (int k = 0; k < 3; ++k) WT[k] = (_Float16*)take((size_t)KDIM * EPAD * (k + 1) * 2);
    _Float16* QENC[3];
    for (int k = 0; k < 3; ++k) QENC[k] = (_Float16*)take((size_t)BATCH * QLEN * KDIM * 2);
    _Float16* DENC[3];
    for (int k = 0; k < 3; ++k) DENC[k] = (_Float16*)take((size_t)BATCH * DLEN * KDIM * 2);
    (void)ws_size; (void)n_in; (void)in_sizes; (void)out_size;

    // 1) gathers
    gather_emb_f16<<<dim3(QALLOC, BATCH), EPAD, 0, stream>>>(query_ids, emb, XQ, QLEN, QALLOC);
    gather_emb_f16<<<dim3(DALLOC, BATCH), EPAD, 0, stream>>>(doc_ids,   emb, XD, DLEN, DALLOC);

    // 2) weight packing
    for (int k = 0; k < 3; ++k) {
        const int total = KDIM * EPAD * (k + 1);
        pack_weights<<<(total + 255) / 256, 256, 0, stream>>>(conv_w[k], WT[k], k + 1);
    }

    // 3) conv + relu + normalize (WMMA implicit GEMM)
    for (int k = 0; k < 3; ++k) {
        const int KPAD = EPAD * (k + 1);
        conv_enc_wmma<<<dim3(QLEN / 16, BATCH), 256, 0, stream>>>(
            XQ, QALLOC, WT[k], KPAD, conv_b[k], QENC[k], QLEN);
        conv_enc_wmma<<<dim3(DLEN / 16, BATCH), 256, 0, stream>>>(
            XD, DALLOC, WT[k], KPAD, conv_b[k], DENC[k], DLEN);
    }

    // 4) 9 matchers (WMMA sim + Gaussian pooling) -> logits
    for (int qi = 0; qi < 3; ++qi) {
        for (int di = 0; di < 3; ++di) {
            const int Qv   = QLEN - qi;     // L - k + 1
            const int Vv   = DLEN - di;
            const int pair = qi * 3 + di;
            matcher_wmma<<<BATCH, 256, 0, stream>>>(
                QENC[qi], DENC[di], query_masks, doc_masks,
                out_logits, Qv, Vv, pair * KNUM);
        }
    }

    // 5) dense head -> scores
    dense_kernel<<<1, 32, 0, stream>>>(out_logits, dense_w, dense_b, out_score);
}
